// MultiHeadAttention_11536282157758
// MI455X (gfx1250) — compile-verified, matrix operand order checked by reading, not run on hardware
//
#include <hip/hip_runtime.h>

// ---------------------------------------------------------------------------
// MultiHeadAttention for MI455X (gfx1250): bf16 WMMA flash-attention pipeline.
// ---------------------------------------------------------------------------

#define KB_ 8
#define KS_ 1024
#define KD_ 1024
#define KH_ 16
#define KDH_ 64
#define KOUT_ 1024

typedef __attribute__((ext_vector_type(16))) __bf16 v16bf;
typedef __attribute__((ext_vector_type(8)))  float  v8f;
typedef __attribute__((ext_vector_type(4)))  int    v4i;

#if defined(__has_builtin)
#if __has_builtin(__builtin_amdgcn_global_load_async_to_lds_b128)
#define HAVE_ASYNC_LDS 1
#endif
#endif
#ifndef HAVE_ASYNC_LDS
#define HAVE_ASYNC_LDS 0
#endif

#if HAVE_ASYNC_LDS
typedef __attribute__((address_space(1))) v4i as1_v4i;
typedef __attribute__((address_space(3))) v4i as3_v4i;
#endif

// Copy 16 bytes global -> LDS. Async path uses the CDNA5 async-DMA
// (ASYNCcnt-tracked) instruction; fallback is a plain b128 load/store.
__device__ inline void copy16_g2l(const void* g, void* l) {
#if HAVE_ASYNC_LDS
  __builtin_amdgcn_global_load_async_to_lds_b128((as1_v4i*)g, (as3_v4i*)l,
                                                 0, 0);
#else
  *(uint4*)l = *(const uint4*)g;
#endif
}

__device__ inline void wait_async_copies() {
#if HAVE_ASYNC_LDS
#if __has_builtin(__builtin_amdgcn_s_wait_asynccnt)
  __builtin_amdgcn_s_wait_asynccnt(0);
#else
  asm volatile("s_wait_asynccnt 0x0" ::: "memory");
#endif
#endif
}

__device__ inline v8f wmma_bf16(v16bf a, v16bf b, v8f c) {
  return __builtin_amdgcn_wmma_f32_16x16x32_bf16(
      /*neg_a=*/false, a, /*neg_b=*/false, b,
      /*c_mod=*/(short)0, c, /*reuse_a=*/false, /*reuse_b=*/false);
}

// Load a 16x32 A-layout bf16 fragment from a row-major source (stride `ld`
// elements). Lane l -> row (l&15); element j -> K = (l>>4)*8+(j/8)*16+(j%8),
// i.e. two runs of 8 contiguous K values -> two 16-byte vector loads.
// Also used for B fragments by pointing it at a transposed (N-major) source.
__device__ inline v16bf load_a16x32(const __bf16* s, int ld, int lane) {
  int r  = lane & 15;
  int kb = (lane >> 4) << 3;
  const __bf16* p = s + r * ld + kb;
  union { uint4 u[2]; v16bf v; } t;
  t.u[0] = *(const uint4*)(p);
  t.u[1] = *(const uint4*)(p + 16);
  return t.v;
}

__device__ inline void store4bf(__bf16* p, float4 f) {
  union { __bf16 h[4]; uint2 u; } t;
  t.h[0] = (__bf16)f.x; t.h[1] = (__bf16)f.y;
  t.h[2] = (__bf16)f.z; t.h[3] = (__bf16)f.w;
  *(uint2*)p = t.u;
}

// ---------------------------------------------------------------------------
// Kernel 1: per-head QKV projection.  grid = B*H*(S/64), block = 256 (8 waves)
// q/k/v written bf16, layout [B,H,S,DH].
// ---------------------------------------------------------------------------
__global__ __launch_bounds__(256) void qkv_kernel(
    const float* __restrict__ x,
    const float* __restrict__ Wq, const float* __restrict__ Wk,
    const float* __restrict__ Wv,
    const float* __restrict__ bq, const float* __restrict__ bk,
    const float* __restrict__ bv,
    __bf16* __restrict__ q, __bf16* __restrict__ k, __bf16* __restrict__ v) {
  __shared__ __align__(16) __bf16 xs[64 * 64];        // x tile  [s][d]
  __shared__ __align__(16) __bf16 wt[3][64 * 64];     // W^T     [e][d]

  int bid  = blockIdx.x;
  int sblk = bid & 15;
  int h    = (bid >> 4) & 15;
  int b    = bid >> 8;
  int tid  = threadIdx.x;
  int s0   = sblk * 64;

  // x tile: vectorized f32 loads, packed bf16 stores.
  const float* xg = x + ((size_t)b * KS_ + s0) * KD_ + h * KDH_;
#pragma unroll
  for (int i = 0; i < 4; ++i) {                       // 1024 float4
    int idx = tid + i * 256;
    int r = idx >> 4, c4 = (idx & 15) << 2;
    float4 f = *(const float4*)(xg + (size_t)r * KD_ + c4);
    store4bf(xs + r * 64 + c4, f);
  }
  // weights: vectorized loads, stored transposed (compile-time m -> the
  // pointer selects fold, keeping global address space).
#pragma unroll
  for (int m = 0; m < 3; ++m) {
    const float* wg = ((m == 0) ? Wq : (m == 1) ? Wk : Wv) + h * KDH_ * KDH_;
#pragma unroll
    for (int i = 0; i < 4; ++i) {                     // 1024 float4
      int idx = tid + i * 256;
      int d = idx >> 4, e4 = (idx & 15) << 2;
      float4 f = *(const float4*)(wg + d * KDH_ + e4);
      wt[m][(e4 + 0) * 64 + d] = (__bf16)f.x;
      wt[m][(e4 + 1) * 64 + d] = (__bf16)f.y;
      wt[m][(e4 + 2) * 64 + d] = (__bf16)f.z;
      wt[m][(e4 + 3) * 64 + d] = (__bf16)f.w;
    }
  }
  __syncthreads();

  int wave = tid >> 5, lane = tid & 31;
  int col = lane & 15, r0 = (lane >> 4) << 3;

#pragma unroll
  for (int m = 0; m < 3; ++m) {
    const float* bg = (m == 0) ? bq : (m == 1) ? bk : bv;
    __bf16* og0 = (m == 0) ? q : (m == 1) ? k : v;
    for (int t = wave; t < 16; t += 8) {              // 4x4 output tiles
      int rt = t >> 2, ct = t & 3;
      v8f acc = {};
#pragma unroll
      for (int kk = 0; kk < 64; kk += 32) {
        v16bf a  = load_a16x32(xs + rt * 16 * 64 + kk, 64, lane);
        v16bf bb = load_a16x32(wt[m] + ct * 16 * 64 + kk, 64, lane);
        acc = wmma_bf16(a, bb, acc);
      }
      float bi = bg[h * KDH_ + ct * 16 + col];
      __bf16* og = og0 +
          ((size_t)(b * KH_ + h) * KS_ + s0 + rt * 16) * KDH_ + ct * 16;
#pragma unroll
      for (int r = 0; r < 8; ++r)
        og[(size_t)(r0 + r) * KDH_ + col] = (__bf16)(acc[r] + bi);
    }
  }
}

// ---------------------------------------------------------------------------
// Kernel 2: flash attention.  grid = B*H*(S/128), block = 256 (8 waves);
// each wave owns 16 query rows and streams over keys in chunks of 32.
// Output o: bf16, layout [B*S, H*DH]  (ready for the aggregation GEMM).
// ---------------------------------------------------------------------------
__global__ __launch_bounds__(256) void attn_kernel(
    const __bf16* __restrict__ q, const __bf16* __restrict__ k,
    const __bf16* __restrict__ v, __bf16* __restrict__ o) {
  __shared__ __align__(16) __bf16 kls[32 * 64];       // K chunk row-major
  __shared__ __align__(16) __bf16 vts[64 * 32];       // V chunk transposed
  __shared__ __align__(16) __bf16 pls[8][16 * 32];    // per-wave P staging

  int bid  = blockIdx.x;
  int qblk = bid & 7;
  int h    = (bid >> 3) & 15;
  int b    = bid >> 7;
  int tid  = threadIdx.x, wave = tid >> 5, lane = tid & 31;
  size_t bh = (size_t)(b * KH_ + h);
  int q0 = qblk * 128 + wave * 16;

  const __bf16* qg = q + (bh * KS_ + q0) * KDH_;
  v16bf qa0 = load_a16x32(qg, KDH_, lane);            // K-dims 0..31
  v16bf qa1 = load_a16x32(qg + 32, KDH_, lane);       // K-dims 32..63

  float mrow[8], lrow[8];
  v8f oacc[4];
#pragma unroll
  for (int r = 0; r < 8; ++r) { mrow[r] = -1e30f; lrow[r] = 0.f; }
#pragma unroll
  for (int n = 0; n < 4; ++n) oacc[n] = (v8f){};

  const float scale = 0.125f;                         // 1/sqrt(64)

  for (int kc = 0; kc < KS_ / 32; ++kc) {
    const __bf16* kg = k + (bh * KS_ + kc * 32) * KDH_;
    const __bf16* vg = v + (bh * KS_ + kc * 32) * KDH_;
    // K chunk is a contiguous 4 KB blob: one async b128 per thread.
    copy16_g2l((const uint4*)kg + tid, (uint4*)kls + tid);
    // V staged transposed: one b128 global load + 8 b16 LDS scatter stores.
    {
      int key = tid >> 3, e0 = (tid & 7) << 3;
      union { uint4 u; __bf16 h8[8]; } t;
      t.u = *(const uint4*)(vg + key * KDH_ + e0);
#pragma unroll
      for (int j = 0; j < 8; ++j) vts[(e0 + j) * 32 + key] = t.h8[j];
    }
    __builtin_prefetch(kg + 32 * KDH_, 0, 0);         // next chunk
    wait_async_copies();
    __syncthreads();

    // scores: S = Q * K^T  (two 16x16 key tiles, K-depth 64)
    v8f s0 = {}, s1 = {};
    {
      v16bf b0  = load_a16x32(kls, 64, lane);
      v16bf b0b = load_a16x32(kls + 32, 64, lane);
      s0 = wmma_bf16(qa0, b0, s0);
      s0 = wmma_bf16(qa1, b0b, s0);
      v16bf b1  = load_a16x32(kls + 16 * 64, 64, lane);
      v16bf b1b = load_a16x32(kls + 16 * 64 + 32, 64, lane);
      s1 = wmma_bf16(qa0, b1, s1);
      s1 = wmma_bf16(qa1, b1b, s1);
    }

    // online softmax update (rows live on fixed lanes of each 16-lane half,
    // so xor-shuffles 1/2/4/8 reduce across exactly one row's columns).
    float alpha[8];
#pragma unroll
    for (int r = 0; r < 8; ++r) {
      s0[r] *= scale; s1[r] *= scale;
      float cm = fmaxf(s0[r], s1[r]);
      cm = fmaxf(cm, __shfl_xor(cm, 1, 32));
      cm = fmaxf(cm, __shfl_xor(cm, 2, 32));
      cm = fmaxf(cm, __shfl_xor(cm, 4, 32));
      cm = fmaxf(cm, __shfl_xor(cm, 8, 32));
      float mn = fmaxf(mrow[r], cm);
      alpha[r] = __expf(mrow[r] - mn);
      float p0 = __expf(s0[r] - mn);
      float p1 = __expf(s1[r] - mn);
      s0[r] = p0; s1[r] = p1;
      float rs = p0 + p1;
      rs += __shfl_xor(rs, 1, 32);
      rs += __shfl_xor(rs, 2, 32);
      rs += __shfl_xor(rs, 4, 32);
      rs += __shfl_xor(rs, 8, 32);
      lrow[r] = lrow[r] * alpha[r] + rs;
      mrow[r] = mn;
    }

    // Re-layout P: C-fragment -> LDS -> A-fragment (bf16).
    {
      __bf16* pp = pls[wave];
      int col = lane & 15, r0 = (lane >> 4) << 3;
#pragma unroll
      for (int r = 0; r < 8; ++r) {
        pp[(r0 + r) * 32 + col]      = (__bf16)s0[r];
        pp[(r0 + r) * 32 + col + 16] = (__bf16)s1[r];
      }
    }
#pragma unroll
    for (int n = 0; n < 4; ++n)
#pragma unroll
      for (int r = 0; r < 8; ++r) oacc[n][r] *= alpha[r];

    v16bf pa = load_a16x32(pls[wave], 32, lane);
#pragma unroll
    for (int n = 0; n < 4; ++n) {
      v16bf bb = load_a16x32(vts + n * 16 * 32, 32, lane);
      oacc[n] = wmma_bf16(pa, bb, oacc[n]);
    }
    __syncthreads();
  }

  // normalize & store bf16 into [B*S, H*DH]
  int col = lane & 15, r0 = (lane >> 4) << 3;
#pragma unroll
  for (int n = 0; n < 4; ++n)
#pragma unroll
    for (int r = 0; r < 8; ++r) {
      float val = oacc[n][r] / lrow[r];
      int row = q0 + r0 + r;
      o[((size_t)b * KS_ + row) * (KH_ * KDH_) + h * KDH_ + n * 16 + col] =
          (__bf16)val;
    }
}

// ---------------------------------------------------------------------------
// Kernel 3: aggregation GEMM  [B*S,1024] x Wa[1024,1024] -> out f32.
// grid = (8192/128)*(1024/64) = 1024, block = 256; block tile 128x64.
// ---------------------------------------------------------------------------
__global__ __launch_bounds__(256) void proj_kernel(
    const __bf16* __restrict__ a, const float* __restrict__ Wa,
    float* __restrict__ out) {
  __shared__ __align__(16) __bf16 als[128 * 32];      // A tile row-major
  __shared__ __align__(16) __bf16 bts[64 * 32];       // Wa tile transposed

  int bid = blockIdx.x;
  int nb = bid & 15, mb = bid >> 4;
  int m0 = mb * 128, n0 = nb * 64;
  int tid = threadIdx.x, wave = tid >> 5, lane = tid & 31;

  v8f acc[4] = {};
  for (int kc = 0; kc < KD_; kc += 32) {
    // A tile: 512 b128 async copies (rows of 32 bf16 = 4 x 16B).
    const __bf16* ag = a + (size_t)m0 * KD_ + kc;
#pragma unroll
    for (int i = 0; i < 2; ++i) {
      int idx = tid + i * 256;
      int r = idx >> 2, c = (idx & 3) << 3;
      copy16_g2l(ag + (size_t)r * KD_ + c, als + r * 32 + c);
    }
    // Wa tile: vector f32 loads, converted + transposed into LDS.
#pragma unroll
    for (int i = 0; i < 2; ++i) {                     // 512 float4
      int idx = tid + i * 256;
      int kk = idx >> 4, n4 = (idx & 15) << 2;
      float4 f = *(const float4*)(Wa + (size_t)(kc + kk) * KOUT_ + n0 + n4);
      bts[(n4 + 0) * 32 + kk] = (__bf16)f.x;
      bts[(n4 + 1) * 32 + kk] = (__bf16)f.y;
      bts[(n4 + 2) * 32 + kk] = (__bf16)f.z;
      bts[(n4 + 3) * 32 + kk] = (__bf16)f.w;
    }
    wait_async_copies();
    __syncthreads();

    v16bf av = load_a16x32(als + wave * 16 * 32, 32, lane);
#pragma unroll
    for (int n = 0; n < 4; ++n) {
      v16bf bb = load_a16x32(bts + n * 16 * 32, 32, lane);
      acc[n] = wmma_bf16(av, bb, acc[n]);
    }
    __syncthreads();
  }

  int col = lane & 15, r0 = (lane >> 4) << 3;
#pragma unroll
  for (int n = 0; n < 4; ++n)
#pragma unroll
    for (int r = 0; r < 8; ++r)
      out[(size_t)(m0 + wave * 16 + r0 + r) * KOUT_ + n0 + n * 16 + col] =
          acc[n][r];
}

// ---------------------------------------------------------------------------
extern "C" void kernel_launch(void* const* d_in, const int* in_sizes, int n_in,
                              void* d_out, int out_size, void* d_ws,
                              size_t ws_size, hipStream_t stream) {
  const float* x  = (const float*)d_in[0];
  const float* Wq = (const float*)d_in[1];
  const float* Wk = (const float*)d_in[2];
  const float* Wv = (const float*)d_in[3];
  const float* bq = (const float*)d_in[4];
  const float* bk = (const float*)d_in[5];
  const float* bv = (const float*)d_in[6];
  const float* Wa = (const float*)d_in[7];
  float* out = (float*)d_out;

  const size_t elems = (size_t)KB_ * KH_ * KS_ * KDH_;  // 8 Mi elements
  __bf16* qb = (__bf16*)d_ws;
  __bf16* kb = qb + elems;
  __bf16* vb = kb + elems;
  __bf16* ob = vb + elems;

  qkv_kernel<<<KB_ * KH_ * (KS_ / 64), 256, 0, stream>>>(
      x, Wq, Wk, Wv, bq, bk, bv, qb, kb, vb);
  attn_kernel<<<KB_ * KH_ * (KS_ / 128), 256, 0, stream>>>(qb, kb, vb, ob);
  proj_kernel<<<(KB_ * KS_ / 128) * (KOUT_ / 64), 256, 0, stream>>>(
      ob, Wa, out);
}
